// StateSpaceBlock_38027640439433
// MI455X (gfx1250) — compile-verified, matrix operand order checked by reading, not run on hardware
//
#include <hip/hip_runtime.h>
#include <hip/hip_bf16.h>

typedef __bf16 bf16;
typedef __attribute__((ext_vector_type(8)))  __bf16 bf16x8;
typedef __attribute__((ext_vector_type(16))) __bf16 bf16x16;
typedef __attribute__((ext_vector_type(8)))  float  f32x8;

// Problem constants (from reference setup_inputs)
#define BB   2
#define LL   1024
#define DM   512
#define DI   1024
#define NS   64
#define KC   4
#define ROWS (BB*LL)          // 2048 token rows

// ---------------------------------------------------------------------------
// f32 -> bf16 convert (weights)
// ---------------------------------------------------------------------------
__global__ __launch_bounds__(256)
void cvt_bf16_kernel(const float* __restrict__ in, bf16* __restrict__ out, int n) {
    int i = blockIdx.x * 256 + threadIdx.x;
    if (i < n) out[i] = (bf16)in[i];
}

// ---------------------------------------------------------------------------
// LayerNorm over d_model=512, one block per token row, write bf16
// ---------------------------------------------------------------------------
__global__ __launch_bounds__(256)
void ln_kernel(const float* __restrict__ x, const float* __restrict__ g,
               const float* __restrict__ b, bf16* __restrict__ xn) {
    const int row = blockIdx.x;                 // 0..2047
    const float* xr = x + (size_t)row * DM;
    const int t = threadIdx.x;
    float v0 = xr[t];
    float v1 = xr[t + 256];

    __shared__ float sred[8];
    __shared__ float smu, srstd;

    float s = v0 + v1;
    for (int off = 16; off; off >>= 1) s += __shfl_xor(s, off, 32);
    if ((t & 31) == 0) sred[t >> 5] = s;
    __syncthreads();
    if (t == 0) {
        float tot = 0.f;
        for (int i = 0; i < 8; ++i) tot += sred[i];
        smu = tot * (1.0f / DM);
    }
    __syncthreads();
    const float mu = smu;
    float d0 = v0 - mu, d1 = v1 - mu;
    float q = d0 * d0 + d1 * d1;
    for (int off = 16; off; off >>= 1) q += __shfl_xor(q, off, 32);
    if ((t & 31) == 0) sred[t >> 5] = q;
    __syncthreads();
    if (t == 0) {
        float tot = 0.f;
        for (int i = 0; i < 8; ++i) tot += sred[i];
        srstd = rsqrtf(tot * (1.0f / DM) + 1e-5f);
    }
    __syncthreads();
    const float rstd = srstd;
    bf16* o = xn + (size_t)row * DM;
    o[t]       = (bf16)(d0 * rstd * g[t]       + b[t]);
    o[t + 256] = (bf16)(d1 * rstd * g[t + 256] + b[t + 256]);
}

// ---------------------------------------------------------------------------
// bf16 WMMA GEMM: C[M,N] = A[M,K] * W[N,K]^T   (f32 accumulate)
// Each wave computes a 32x32 output block: 2x2 tiles of 16x16, 4 accumulators.
// Double-buffered fragment loads hide L2 latency behind the 4 WMMAs.
// Requires M, N multiples of 32 and K multiple of 32 (all true here).
// ---------------------------------------------------------------------------
__global__ __launch_bounds__(256)
void wmma_gemm_bf16(const bf16* __restrict__ A, const bf16* __restrict__ W,
                    float* __restrict__ C, int M, int N, int Kdim) {
    const int lane = threadIdx.x & 31;
    const int wave = threadIdx.x >> 5;
    const int stN = N >> 5;                      // 32-wide supertiles
    const int stM = M >> 5;
    const int st  = blockIdx.x * 8 + wave;
    if (st >= stM * stN) return;
    const int tm = (st / stN) << 5;              // row base of 32x32 block
    const int tn = (st % stN) << 5;              // col base of 32x32 block

    // 16-bit A/B fragment layout: lane&15 = row, lane>>4 selects K half (0/8),
    // VGPRs 0..3 hold K=khalf..+7, VGPRs 4..7 hold K=khalf+16..+23.
    const int row   = lane & 15;
    const int khalf = (lane >> 4) << 3;
    const bf16* a0p = A + (size_t)(tm + row)      * Kdim + khalf;
    const bf16* a1p = A + (size_t)(tm + 16 + row) * Kdim + khalf;
    const bf16* b0p = W + (size_t)(tn + row)      * Kdim + khalf;
    const bf16* b1p = W + (size_t)(tn + 16 + row) * Kdim + khalf;

    auto loadfrag = [](const bf16* p, int kk) -> bf16x16 {
        bf16x8 lo = *(const bf16x8*)(p + kk);        // global_load_b128
        bf16x8 hi = *(const bf16x8*)(p + kk + 16);   // global_load_b128
        bf16x16 v;
#pragma unroll
        for (int i = 0; i < 8; ++i) { v[i] = lo[i]; v[i + 8] = hi[i]; }
        return v;
    };

    f32x8 acc00 = {0.f,0.f,0.f,0.f,0.f,0.f,0.f,0.f};
    f32x8 acc01 = acc00, acc10 = acc00, acc11 = acc00;

    bf16x16 a0 = loadfrag(a0p, 0);
    bf16x16 a1 = loadfrag(a1p, 0);
    bf16x16 b0 = loadfrag(b0p, 0);
    bf16x16 b1 = loadfrag(b1p, 0);

    for (int kk = 32; kk < Kdim; kk += 32) {
        // prefetch one chunk further ahead (L2 -> WGP$)
        if (kk + 32 < Kdim) {
            __builtin_prefetch(a0p + kk + 32, 0, 3);
            __builtin_prefetch(b0p + kk + 32, 0, 3);
        }
        // issue next-iteration fragment loads before consuming current ones
        bf16x16 na0 = loadfrag(a0p, kk);
        bf16x16 na1 = loadfrag(a1p, kk);
        bf16x16 nb0 = loadfrag(b0p, kk);
        bf16x16 nb1 = loadfrag(b1p, kk);

        acc00 = __builtin_amdgcn_wmma_f32_16x16x32_bf16(false, a0, false, b0, (short)0, acc00, false, false);
        acc01 = __builtin_amdgcn_wmma_f32_16x16x32_bf16(false, a0, false, b1, (short)0, acc01, false, false);
        acc10 = __builtin_amdgcn_wmma_f32_16x16x32_bf16(false, a1, false, b0, (short)0, acc10, false, false);
        acc11 = __builtin_amdgcn_wmma_f32_16x16x32_bf16(false, a1, false, b1, (short)0, acc11, false, false);

        a0 = na0; a1 = na1; b0 = nb0; b1 = nb1;
    }
    // epilogue: last K-step
    acc00 = __builtin_amdgcn_wmma_f32_16x16x32_bf16(false, a0, false, b0, (short)0, acc00, false, false);
    acc01 = __builtin_amdgcn_wmma_f32_16x16x32_bf16(false, a0, false, b1, (short)0, acc01, false, false);
    acc10 = __builtin_amdgcn_wmma_f32_16x16x32_bf16(false, a1, false, b0, (short)0, acc10, false, false);
    acc11 = __builtin_amdgcn_wmma_f32_16x16x32_bf16(false, a1, false, b1, (short)0, acc11, false, false);

    // C/D layout: lane&15 = N column; lane>>4 selects M 0..7 vs 8..15; VGPR r = M row.
    const int nc = lane & 15;
    const int mb = (lane >> 4) << 3;
    float* c00 = C + (size_t)(tm + mb) * N + tn + nc;
    float* c10 = C + (size_t)(tm + 16 + mb) * N + tn + nc;
#pragma unroll
    for (int r = 0; r < 8; ++r) {
        c00[(size_t)r * N]      = acc00[r];
        c00[(size_t)r * N + 16] = acc01[r];
        c10[(size_t)r * N]      = acc10[r];
        c10[(size_t)r * N + 16] = acc11[r];
    }
}

// ---------------------------------------------------------------------------
// Causal depthwise conv1d (K=4) + bias + SiLU; input x_proj = xz[:, :, 0:DI]
// ---------------------------------------------------------------------------
__global__ __launch_bounds__(256)
void conv_silu_kernel(const float* __restrict__ xz, const float* __restrict__ Wconv,
                      const float* __restrict__ bconv,
                      bf16* __restrict__ xc_bf, float* __restrict__ xc_f) {
    int i = blockIdx.x * 256 + threadIdx.x;
    if (i >= BB * LL * DI) return;
    const int c = i & (DI - 1);
    const int l = (i >> 10) & (LL - 1);
    const int b = i >> 20;
    float acc = bconv[c];
#pragma unroll
    for (int k = 0; k < KC; ++k) {
        const int ls = l - (KC - 1) + k;
        if (ls >= 0)
            acc += Wconv[c * KC + k] * xz[((size_t)b * LL + ls) * (2 * DI) + c];
    }
    const float sv = acc / (1.f + __expf(-acc));
    xc_f[i]  = sv;
    xc_bf[i] = (bf16)sv;
}

// ---------------------------------------------------------------------------
// dt epilogue: dt = softplus(dt_pre + bdt[c])   (in place)
// ---------------------------------------------------------------------------
__global__ __launch_bounds__(256)
void softplus_bias_kernel(float* __restrict__ dt, const float* __restrict__ bdt, int n) {
    int i = blockIdx.x * 256 + threadIdx.x;
    if (i >= n) return;
    const float v = dt[i] + bdt[i & (DI - 1)];
    dt[i] = (v > 20.f) ? v : log1pf(__expf(v));
}

// ---------------------------------------------------------------------------
// Selective scan: one wave32 per (b, d) channel; 2 of 64 states per lane.
// Fuses  y + u*D  and the SiLU(z) gate; writes bf16 GEMM input.
// ---------------------------------------------------------------------------
__global__ __launch_bounds__(256)
void scan_kernel(const float* __restrict__ dt, const float* __restrict__ BC,
                 const float* __restrict__ xc, const float* __restrict__ xz,
                 const float* __restrict__ A_log, const float* __restrict__ Dvec,
                 bf16* __restrict__ ygate) {
    const int lane = threadIdx.x & 31;
    const int chan = blockIdx.x * 8 + (threadIdx.x >> 5);   // 0..2047
    const int b = chan >> 10;
    const int d = chan & (DI - 1);
    const int n0 = lane, n1 = lane + 32;

    const float A0 = -__expf(A_log[d * NS + n0]);
    const float A1 = -__expf(A_log[d * NS + n1]);
    const float Dv = Dvec[d];

    float h0 = 0.f, h1 = 0.f;
    const size_t lbase = (size_t)b * LL;

    for (int l = 0; l < LL; ++l) {
        const size_t rc = lbase + l;
        const float dtv = dt[rc * DI + d];      // broadcast within wave
        const float u   = xc[rc * DI + d];
        const float Bv0 = BC[rc * (2 * NS) + n0];
        const float Bv1 = BC[rc * (2 * NS) + n1];
        const float Cv0 = BC[rc * (2 * NS) + NS + n0];
        const float Cv1 = BC[rc * (2 * NS) + NS + n1];

        h0 = __expf(dtv * A0) * h0 + dtv * Bv0 * u;
        h1 = __expf(dtv * A1) * h1 + dtv * Bv1 * u;

        float p = h0 * Cv0 + h1 * Cv1;
        for (int off = 16; off; off >>= 1) p += __shfl_xor(p, off, 32);

        if (lane == 0) {
            const float y = p + u * Dv;
            const float z = xz[rc * (2 * DI) + DI + d];
            const float gate = z / (1.f + __expf(-z));
            ygate[rc * DI + d] = (bf16)(y * gate);
        }
    }
}

// ---------------------------------------------------------------------------
// out = residual + outtmp
// ---------------------------------------------------------------------------
__global__ __launch_bounds__(256)
void residual_add_kernel(const float* __restrict__ x, const float* __restrict__ t,
                         float* __restrict__ out, int n) {
    int i = blockIdx.x * 256 + threadIdx.x;
    if (i < n) out[i] = x[i] + t[i];
}

// ---------------------------------------------------------------------------
// launch
// ---------------------------------------------------------------------------
static inline size_t alignup(size_t v) { return (v + 255) & ~(size_t)255; }

extern "C" void kernel_launch(void* const* d_in, const int* in_sizes, int n_in,
                              void* d_out, int out_size, void* d_ws, size_t ws_size,
                              hipStream_t stream) {
    const float* x     = (const float*)d_in[0];
    const float* ln_g  = (const float*)d_in[1];
    const float* ln_b  = (const float*)d_in[2];
    const float* Win   = (const float*)d_in[3];
    const float* Wconv = (const float*)d_in[4];
    const float* bconv = (const float*)d_in[5];
    const float* Wx    = (const float*)d_in[6];
    const float* Wdt   = (const float*)d_in[7];
    const float* bdt   = (const float*)d_in[8];
    const float* A_log = (const float*)d_in[9];
    const float* Dvec  = (const float*)d_in[10];
    const float* Wout  = (const float*)d_in[11];
    float* out = (float*)d_out;

    // Workspace layout
    char* p = (char*)d_ws;
    size_t off = 0;
    auto take = [&](size_t bytes) { void* r = p + off; off = alignup(off + bytes); return r; };

    bf16*  xn_bf   = (bf16*) take((size_t)ROWS * DM * 2);            // 2 MB
    bf16*  Win_bf  = (bf16*) take((size_t)2 * DI * DM * 2);          // 2 MB
    bf16*  Wdt_bf  = (bf16*) take((size_t)DI * DI * 2);              // 2 MB
    bf16*  Wx_bf   = (bf16*) take((size_t)2 * NS * DI * 2);          // 256 KB
    bf16*  Wout_bf = (bf16*) take((size_t)DM * DI * 2);              // 1 MB
    float* xz      = (float*)take((size_t)ROWS * 2 * DI * 4);        // 16 MB
    bf16*  xc_bf   = (bf16*) take((size_t)ROWS * DI * 2);            // 4 MB
    float* xc_f    = (float*)take((size_t)ROWS * DI * 4);            // 8 MB
    float* dt      = (float*)take((size_t)ROWS * DI * 4);            // 8 MB
    float* BC      = (float*)take((size_t)ROWS * 2 * NS * 4);        // 1 MB
    bf16*  ygate   = (bf16*) take((size_t)ROWS * DI * 2);            // 4 MB
    float* outtmp  = (float*)take((size_t)ROWS * DM * 4);            // 4 MB
    (void)ws_size; (void)n_in; (void)in_sizes; (void)out_size;

    // 1) weight conversions f32 -> bf16
    cvt_bf16_kernel<<<(2*DI*DM + 255)/256, 256, 0, stream>>>(Win,  Win_bf,  2*DI*DM);
    cvt_bf16_kernel<<<(DI*DI   + 255)/256, 256, 0, stream>>>(Wdt,  Wdt_bf,  DI*DI);
    cvt_bf16_kernel<<<(2*NS*DI + 255)/256, 256, 0, stream>>>(Wx,   Wx_bf,   2*NS*DI);
    cvt_bf16_kernel<<<(DM*DI   + 255)/256, 256, 0, stream>>>(Wout, Wout_bf, DM*DI);

    // 2) LayerNorm -> bf16
    ln_kernel<<<ROWS, 256, 0, stream>>>(x, ln_g, ln_b, xn_bf);

    // 3) xz = xn @ Win^T      M=2048 N=2048 K=512
    {
        int st = (ROWS/32) * ((2*DI)/32);
        wmma_gemm_bf16<<<(st + 7)/8, 256, 0, stream>>>(xn_bf, Win_bf, xz, ROWS, 2*DI, DM);
    }

    // 4) causal depthwise conv + SiLU
    conv_silu_kernel<<<(BB*LL*DI + 255)/256, 256, 0, stream>>>(xz, Wconv, bconv, xc_bf, xc_f);

    // 5) dt_pre = xc @ Wdt^T  M=2048 N=1024 K=1024 ; then softplus(+bias)
    {
        int st = (ROWS/32) * (DI/32);
        wmma_gemm_bf16<<<(st + 7)/8, 256, 0, stream>>>(xc_bf, Wdt_bf, dt, ROWS, DI, DI);
    }
    softplus_bias_kernel<<<(ROWS*DI + 255)/256, 256, 0, stream>>>(dt, bdt, ROWS*DI);

    // 6) BC = xc @ Wx^T       M=2048 N=128 K=1024
    {
        int st = (ROWS/32) * ((2*NS)/32);
        wmma_gemm_bf16<<<(st + 7)/8, 256, 0, stream>>>(xc_bf, Wx_bf, BC, ROWS, 2*NS, DI);
    }

    // 7) selective scan + D-term + SiLU gate (one wave per (b,d) channel)
    scan_kernel<<<(BB*DI)/8, 256, 0, stream>>>(dt, BC, xc_f, xz, A_log, Dvec, ygate);

    // 8) outtmp = ygate @ Wout^T   M=2048 N=512 K=1024
    {
        int st = (ROWS/32) * (DM/32);
        wmma_gemm_bf16<<<(st + 7)/8, 256, 0, stream>>>(ygate, Wout_bf, outtmp, ROWS, DM, DI);
    }

    // 9) out = residual + outtmp
    residual_add_kernel<<<(ROWS*DM + 255)/256, 256, 0, stream>>>(x, outtmp, out, ROWS*DM);
}
